// FlaxSapamaAttention_47287589929585
// MI455X (gfx1250) — matrix operand order repaired and checked by reading, not compiled, not run on hardware
//
#include <hip/hip_runtime.h>
#include <hip/hip_bf16.h>

typedef __attribute__((ext_vector_type(16))) __bf16 v16bf;
typedef __attribute__((ext_vector_type(8)))  __bf16 v8bf;
typedef __attribute__((ext_vector_type(8)))  float  v8f;
typedef __attribute__((ext_vector_type(4)))  unsigned int v4u;
typedef __attribute__((ext_vector_type(8)))  int v8i;
typedef __attribute__((ext_vector_type(4)))  int v4i;

#define NHEADS   32
#define NKVHEADS 8
#define HEADDIM  128
#define HIDDEN   4096
#define BATCH    2
#define SEQ      2048

#if defined(__has_builtin)
#  if __has_builtin(__builtin_amdgcn_tensor_load_to_lds) && __has_builtin(__builtin_amdgcn_s_wait_tensorcnt)
#    define HAVE_TDM 1
#  endif
#endif
#ifndef HAVE_TDM
#  define HAVE_TDM 0
#endif

// ---------- bf16 <-> f32 helpers (bit-level) ----------
__device__ __forceinline__ unsigned short f2bf_bits(float f) {
  union { float f; unsigned u; } v; v.f = f;
  unsigned u = v.u;
  if ((u & 0x7fffffffu) > 0x7f800000u) return (unsigned short)((u >> 16) | 0x0040u); // NaN
  return (unsigned short)((u + 0x7fffu + ((u >> 16) & 1u)) >> 16);                   // RNE
}
__device__ __forceinline__ __bf16 f2bf(float f) {
  unsigned short s = f2bf_bits(f);
  return *reinterpret_cast<__bf16*>(&s);
}
__device__ __forceinline__ float bf2f(__bf16 x) {
  unsigned short s = *reinterpret_cast<unsigned short*>(&x);
  union { unsigned u; float f; } v; v.u = ((unsigned)s) << 16;
  return v.f;
}

// ---------- WMMA fragment builders (ISA 7.12.2 layouts, wave32) ----------
// A 16x32 bf16: lane m = lane&15, element e -> K = (e<8 ? e : e+8) + 8*half
__device__ __forceinline__ v16bf frag_a_rows(const __bf16* rowbase, int hl) {
  v8bf lo = *(const v8bf*)(rowbase + 8 * hl);
  v8bf hi = *(const v8bf*)(rowbase + 16 + 8 * hl);
  v16bf r;
#pragma unroll
  for (int i = 0; i < 8; ++i) { r[i] = lo[i]; r[i + 8] = hi[i]; }
  return r;
}
// B 32x16 bf16 stored as [N][K] rows: lane n = lane&15, element e -> K = (e<8 ? e : e+8) + 8*half
__device__ __forceinline__ v16bf frag_b_rows(const __bf16* rowbase, int hl) {
  v8bf lo = *(const v8bf*)(rowbase + 8 * hl);
  v8bf hi = *(const v8bf*)(rowbase + 16 + 8 * hl);
  v16bf r;
#pragma unroll
  for (int i = 0; i < 8; ++i) { r[i] = lo[i]; r[i + 8] = hi[i]; }
  return r;
}
__device__ __forceinline__ v8f wmma_bf16(v16bf a, v16bf b, v8f c) {
  return __builtin_amdgcn_wmma_f32_16x16x32_bf16(false, a, false, b, (short)0, c, false, false);
}

// ---------- Tensor Data Mover: 2D tile load Global -> LDS (ISA ch.8 D#) ----------
#if HAVE_TDM
__device__ __forceinline__ void tdm_load_2d(unsigned lds_addr, const void* gptr,
                                            unsigned tile_w /*elems*/, unsigned tile_h,
                                            unsigned long long row_stride /*elems*/,
                                            unsigned pad_interval_code, unsigned pad_amount_code) {
  unsigned long long ga = (unsigned long long)gptr;
  v4u g0;
  g0[0] = 1u;                                               // count=1, user descriptor
  g0[1] = lds_addr;                                         // LDS byte address
  g0[2] = (unsigned)(ga & 0xffffffffu);                     // global_addr[31:0]
  g0[3] = (unsigned)((ga >> 32) & 0x01ffffffu) | (2u << 30);// global_addr[56:32] | type=2
  const unsigned td0 = 0x40000000u, td1 = 0x40000000u;      // huge tensor dims (no clipping)
  v8i g1;
  g1[0] = (int)((1u << 16)                                  // data_size: 2 bytes
              | (1u << 20)                                  // pad_enable
              | (pad_interval_code << 22)
              | (pad_amount_code << 25));
  g1[1] = (int)((td0 & 0xffffu) << 16);                     // atomic_barrier=0 | tensor_dim0[15:0]
  g1[2] = (int)((td0 >> 16) | ((td1 & 0xffffu) << 16));     // tensor_dim0[31:16] | tensor_dim1[15:0]
  g1[3] = (int)((td1 >> 16) | (tile_w << 16));              // tensor_dim1[31:16] | tile_dim0
  g1[4] = (int)(tile_h & 0xffffu);                          // tile_dim1 | tile_dim2=0
  g1[5] = (int)(unsigned)(row_stride & 0xffffffffu);        // tensor_dim0_stride[31:0]
  g1[6] = (int)(unsigned)((row_stride >> 32) & 0xffffu);    // stride[47:32] | dim1_stride lo = 0
  g1[7] = 0;
  v4i g2 = {0, 0, 0, 0};
  v4i g3 = {0, 0, 0, 0};
#if defined(__clang_major__) && (__clang_major__ >= 23)
  v8i gz = {0, 0, 0, 0, 0, 0, 0, 0};
  __builtin_amdgcn_tensor_load_to_lds(g0, g1, g2, g3, gz, 0);
#else
  __builtin_amdgcn_tensor_load_to_lds(g0, g1, g2, g3, 0);
#endif
}
#endif

// ---------- elementwise fp32 -> bf16 ----------
__global__ __launch_bounds__(256) void k_f32_to_bf16(const float* __restrict__ in,
                                                     __bf16* __restrict__ out, size_t n) {
  size_t i = ((size_t)blockIdx.x * 256 + threadIdx.x) * 4;
  if (i + 3 < n) {
    float4 v = *(const float4*)(in + i);
    ushort4 r;
    r.x = f2bf_bits(v.x); r.y = f2bf_bits(v.y); r.z = f2bf_bits(v.z); r.w = f2bf_bits(v.w);
    *(ushort4*)(out + i) = r;
  } else {
    for (; i < n; ++i) out[i] = f2bf(in[i]);
  }
}

// ---------- tiled transpose + convert: W[K][N] f32 -> WT[N][K] bf16 ----------
__global__ __launch_bounds__(256) void k_transpose_w(const float* __restrict__ W,
                                                     __bf16* __restrict__ WT, int K, int N) {
  __shared__ float tile[32][33];
  int n0 = blockIdx.x * 32, k0 = blockIdx.y * 32;
  int tx = threadIdx.x & 31, ty = threadIdx.x >> 5; // 32 x 8
#pragma unroll
  for (int i = 0; i < 4; ++i)
    tile[ty + i * 8][tx] = W[(size_t)(k0 + ty + i * 8) * N + n0 + tx];
  __syncthreads();
#pragma unroll
  for (int i = 0; i < 4; ++i)
    WT[(size_t)(n0 + ty + i * 8) * K + k0 + tx] = f2bf(tile[tx][ty + i * 8]);
}

// ---------- WMMA GEMM: C[M,N] = A[M,K](bf16) * BT[N,K](bf16)^T ----------
// Block tile 128x128, 8 waves each own a 16-row strip; BK=32, double-buffered LDS fed by TDM.
__device__ __forceinline__ void gemm_stage_manual(__bf16 (*dstA)[40], __bf16 (*dstB)[40],
                                                  const __bf16* A, const __bf16* BT,
                                                  int m0, int n0, int K, int k0, int tid) {
  int r = tid >> 1, c = (tid & 1) * 16;
  *(v8bf*)&dstA[r][c]     = *(const v8bf*)&A[(size_t)(m0 + r) * K + k0 + c];
  *(v8bf*)&dstA[r][c + 8] = *(const v8bf*)&A[(size_t)(m0 + r) * K + k0 + c + 8];
  *(v8bf*)&dstB[r][c]     = *(const v8bf*)&BT[(size_t)(n0 + r) * K + k0 + c];
  *(v8bf*)&dstB[r][c + 8] = *(const v8bf*)&BT[(size_t)(n0 + r) * K + k0 + c + 8];
}

template <bool OUT_BF16>
__global__ __launch_bounds__(256) void k_gemm_bf16(const __bf16* __restrict__ A,
                                                   const __bf16* __restrict__ BT,
                                                   void* __restrict__ C, int M, int N, int K) {
  __shared__ __attribute__((aligned(16))) __bf16 As[2][128][40];  // 32 cols + 8 pad (TDM pads 16B/64B)
  __shared__ __attribute__((aligned(16))) __bf16 Bs[2][128][40];
  const int m0 = blockIdx.y * 128;
  const int n0 = blockIdx.x * 128;
  const int tid = threadIdx.x;
  const int wave = tid >> 5, lane = tid & 31, hl = lane >> 4, ln = lane & 15;
  v8f acc[8] = {};
  const int nsteps = K >> 5;

  // stage 0
#if HAVE_TDM
  if (wave == 0) {
    tdm_load_2d((unsigned)(size_t)&As[0][0][0], A + (size_t)m0 * K, 32, 128, (unsigned long long)K, 3u, 3u);
    tdm_load_2d((unsigned)(size_t)&Bs[0][0][0], BT + (size_t)n0 * K, 32, 128, (unsigned long long)K, 3u, 3u);
  }
#else
  gemm_stage_manual(As[0], Bs[0], A, BT, m0, n0, K, 0, tid);
#endif

  for (int i = 0; i < nsteps; ++i) {
    const int cur = i & 1;
    __syncthreads();  // everyone done reading buf[cur^1] from previous iteration
    if (i + 1 < nsteps) {
      const int k1 = (i + 1) << 5;
#if HAVE_TDM
      if (wave == 0) {
        tdm_load_2d((unsigned)(size_t)&As[cur ^ 1][0][0], A + (size_t)m0 * K + k1, 32, 128, (unsigned long long)K, 3u, 3u);
        tdm_load_2d((unsigned)(size_t)&Bs[cur ^ 1][0][0], BT + (size_t)n0 * K + k1, 32, 128, (unsigned long long)K, 3u, 3u);
        __builtin_amdgcn_s_wait_tensorcnt(2);  // stage i done; stage i+1 in flight
      }
#else
      gemm_stage_manual(As[cur ^ 1], Bs[cur ^ 1], A, BT, m0, n0, K, k1, tid);
#endif
    }
#if HAVE_TDM
    else if (wave == 0) {
      __builtin_amdgcn_s_wait_tensorcnt(0);
    }
#endif
    __syncthreads();  // stage i visible to all waves

    v16bf af = frag_a_rows(&As[cur][wave * 16 + ln][0], hl);
#pragma unroll
    for (int g = 0; g < 8; ++g) {
      v16bf bf = frag_b_rows(&Bs[cur][g * 16 + ln][0], hl);
      acc[g] = wmma_bf16(af, bf, acc[g]);
    }
  }

#pragma unroll
  for (int g = 0; g < 8; ++g)
#pragma unroll
    for (int r = 0; r < 8; ++r) {
      int row = m0 + wave * 16 + r + 8 * hl;
      int col = n0 + g * 16 + ln;
      if (OUT_BF16) ((__bf16*)C)[(size_t)row * N + col] = f2bf(acc[g][r]);
      else          ((float*)C)[(size_t)row * N + col] = acc[g][r];
    }
}

// ---------- RoPE in-place on bf16 [tok][heads][128] ----------
__global__ __launch_bounds__(256) void k_rope(__bf16* __restrict__ X,
                                              const int* __restrict__ pos_ids,
                                              int hshift, size_t total) {
  size_t t = (size_t)blockIdx.x * 256 + threadIdx.x;
  if (t >= total) return;
  int i = (int)(t & 63);
  size_t rowh = t >> 6;                 // (b*S+s)*nheads + h
  size_t tok = rowh >> hshift;          // b*S+s
  __bf16* base = X + rowh * HEADDIM;
  float p = (float)pos_ids[tok];
  float ang = p * __expf(-0.14391156516f * (float)i);  // p * 10000^(-i/64)
  float c, s;
  __sincosf(ang, &s, &c);
  float x0 = bf2f(base[i]);
  float x1 = bf2f(base[i + 64]);
  base[i]      = f2bf(x0 * c - x1 * s);
  base[i + 64] = f2bf(x1 * c + x0 * s);
}

// ---------- flash attention: grid (S/64, NHEADS, B), block 128 (4 waves) ----------
__global__ __launch_bounds__(128) void k_attn(const __bf16* __restrict__ Q,
                                              const __bf16* __restrict__ Kc,
                                              const __bf16* __restrict__ V,
                                              const float* __restrict__ amask,
                                              __bf16* __restrict__ O) {
  const int qb = blockIdx.x * 64;
  const int h  = blockIdx.y;
  const int b  = blockIdx.z;
  const int kvh = h >> 2;                     // N_GROUPS = 4
  const int tid = threadIdx.x;
  const int wave = tid >> 5, lane = tid & 31, hl = lane >> 4, ln = lane & 15;
  const float scale = 0.088388347648318447f;  // 1/sqrt(128)

  __shared__ __attribute__((aligned(16))) __bf16 Ks[32][136];   // 32 keys x 128 dims (+8 pad = TDM 16B pad / 256B)
  __shared__ __attribute__((aligned(16))) __bf16 Vt[128][40];   // transposed V: dim x 32 keys (+8 pad)
  __shared__ __attribute__((aligned(16))) __bf16 Ps[4][16][40]; // per-wave P staging 16x32 (+8 pad)

  const __bf16* Qrow = Q + (((size_t)b * SEQ + qb + wave * 16 + ln) * NHEADS + h) * HEADDIM;
  v16bf qf[4];
#pragma unroll
  for (int f = 0; f < 4; ++f) qf[f] = frag_a_rows(Qrow + f * 32, hl);

  v8f acc[8] = {};
  float mrow[8], lrow[8];
#pragma unroll
  for (int r = 0; r < 8; ++r) { mrow[r] = -1e30f; lrow[r] = 0.0f; }

  const int nkb = qb / 32 + 2;  // causal: keys up to qb+63
  for (int kb = 0; kb < nkb; ++kb) {
    const int kbase = kb * 32;
    __syncthreads();  // protect previous iteration's K/V reads
#if HAVE_TDM
    if (wave == 0)
      tdm_load_2d((unsigned)(size_t)&Ks[0][0],
                  Kc + (((size_t)b * SEQ + kbase) * NKVHEADS + kvh) * HEADDIM,
                  128, 32, (unsigned long long)(NKVHEADS * HEADDIM), 5u, 3u);
#endif
    {
      int r = tid >> 2;
      int c0 = (tid & 3) * 32;
#if !HAVE_TDM
      const __bf16* ksrc = Kc + (((size_t)b * SEQ + kbase + r) * NKVHEADS + kvh) * HEADDIM + c0;
#pragma unroll
      for (int u = 0; u < 4; ++u)
        *(v8bf*)&Ks[r][c0 + u * 8] = *(const v8bf*)(ksrc + u * 8);
#endif
      const __bf16* vsrc = V + (((size_t)b * SEQ + kbase + r) * NKVHEADS + kvh) * HEADDIM + c0;
#pragma unroll
      for (int u4 = 0; u4 < 4; ++u4) {
        v8bf vv = *(const v8bf*)(vsrc + u4 * 8);
#pragma unroll
        for (int u = 0; u < 8; ++u) Vt[c0 + u4 * 8 + u][r] = vv[u];
      }
    }
#if HAVE_TDM
    if (wave == 0) __builtin_amdgcn_s_wait_tensorcnt(0);
#endif
    __syncthreads();

    // S = Q * K^T  (two 16x16 tiles, dim reduced in 4 chunks of 32)
    v8f s0 = {}, s1 = {};
#pragma unroll
    for (int f = 0; f < 4; ++f) {
      v16bf b0 = frag_b_rows(&Ks[ln][f * 32], hl);
      v16bf b1 = frag_b_rows(&Ks[16 + ln][f * 32], hl);
      s0 = wmma_bf16(qf[f], b0, s0);
      s1 = wmma_bf16(qf[f], b1, s1);
    }

    // scale + causal/pad bias, online softmax
    const float am0 = amask[(size_t)b * SEQ + kbase + ln];
    const float am1 = amask[(size_t)b * SEQ + kbase + 16 + ln];
    float tmax[8], rs[8];
#pragma unroll
    for (int r = 0; r < 8; ++r) {
      int qi = qb + wave * 16 + r + 8 * hl;
      float v0 = s0[r] * scale + ((kbase + ln <= qi && am0 > 0.0f) ? 0.0f : -1e30f);
      float v1 = s1[r] * scale + ((kbase + 16 + ln <= qi && am1 > 0.0f) ? 0.0f : -1e30f);
      s0[r] = v0; s1[r] = v1;
      tmax[r] = fmaxf(v0, v1);
    }
#pragma unroll
    for (int m = 1; m < 16; m <<= 1)
#pragma unroll
      for (int r = 0; r < 8; ++r) tmax[r] = fmaxf(tmax[r], __shfl_xor(tmax[r], m));
    float alpha[8];
#pragma unroll
    for (int r = 0; r < 8; ++r) {
      float nm = fmaxf(mrow[r], tmax[r]);
      alpha[r] = __expf(mrow[r] - nm);
      mrow[r] = nm;
      float p0 = __expf(s0[r] - nm);
      float p1 = __expf(s1[r] - nm);
      s0[r] = p0; s1[r] = p1;
      rs[r] = p0 + p1;
    }
#pragma unroll
    for (int m = 1; m < 16; m <<= 1)
#pragma unroll
      for (int r = 0; r < 8; ++r) rs[r] += __shfl_xor(rs[r], m);
#pragma unroll
    for (int r = 0; r < 8; ++r) lrow[r] = lrow[r] * alpha[r] + rs[r];
#pragma unroll
    for (int g = 0; g < 8; ++g)
#pragma unroll
      for (int r = 0; r < 8; ++r) acc[g][r] *= alpha[r];

    // stage P (C-layout -> A-layout via per-wave LDS)
#pragma unroll
    for (int r = 0; r < 8; ++r) {
      Ps[wave][r + 8 * hl][ln]      = f2bf(s0[r]);
      Ps[wave][r + 8 * hl][16 + ln] = f2bf(s1[r]);
    }
    v16bf pf = frag_a_rows(&Ps[wave][ln][0], hl);
#pragma unroll
    for (int g = 0; g < 8; ++g) {
      v16bf vb = frag_b_rows(&Vt[g * 16 + ln][0], hl);
      acc[g] = wmma_bf16(pf, vb, acc[g]);
    }
  }

  // normalize + write [tok][h*128+d] as bf16
#pragma unroll
  for (int r = 0; r < 8; ++r) {
    float inv = 1.0f / lrow[r];
    int qi = qb + wave * 16 + r + 8 * hl;
    __bf16* dst = O + (((size_t)b * SEQ + qi) * NHEADS + h) * HEADDIM + ln;
#pragma unroll
    for (int g = 0; g < 8; ++g) dst[g * 16] = f2bf(acc[g][r] * inv);
  }
}

// ---------- host ----------
extern "C" void kernel_launch(void* const* d_in, const int* in_sizes, int n_in,
                              void* d_out, int out_size, void* d_ws, size_t ws_size,
                              hipStream_t stream) {
  (void)in_sizes; (void)n_in; (void)out_size; (void)ws_size;
  const float* hs    = (const float*)d_in[0];
  const float* amask = (const float*)d_in[1];
  const int*   pos   = (const int*)d_in[2];
  const float* Wq    = (const float*)d_in[3];
  const float* Wk    = (const float*)d_in[4];
  const float* Wv    = (const float*)d_in[5];
  const float* Wo    = (const float*)d_in[6];

  const int M = BATCH * SEQ;            // 4096 tokens
  char* w = (char*)d_ws;
  size_t off = 0;
  auto carve = [&](size_t elems) {
    __bf16* p = (__bf16*)(w + off);
    off += ((elems * sizeof(__bf16) + 255) / 256) * 256;
    return p;
  };
  __bf16* Xb  = carve((size_t)M * HIDDEN);
  __bf16* WqT = carve((size_t)HIDDEN * HIDDEN);
  __bf16* WkT = carve((size_t)(NKVHEADS * HEADDIM) * HIDDEN);
  __bf16* WvT = carve((size_t)(NKVHEADS * HEADDIM) * HIDDEN);
  __bf16* WoT = carve((size_t)HIDDEN * HIDDEN);
  __bf16* Qb  = carve((size_t)M * HIDDEN);
  __bf16* Kb  = carve((size_t)M * NKVHEADS * HEADDIM);
  __bf16* Vb  = carve((size_t)M * NKVHEADS * HEADDIM);
  __bf16* Ab  = carve((size_t)M * HIDDEN);

  // 1) hidden -> bf16
  k_f32_to_bf16<<<dim3(((size_t)M * HIDDEN) / 1024), 256, 0, stream>>>(hs, Xb, (size_t)M * HIDDEN);

  // 2) weights -> bf16 transposed [N][K]
  k_transpose_w<<<dim3(HIDDEN / 32, HIDDEN / 32), 256, 0, stream>>>(Wq, WqT, HIDDEN, HIDDEN);
  k_transpose_w<<<dim3((NKVHEADS * HEADDIM) / 32, HIDDEN / 32), 256, 0, stream>>>(Wk, WkT, HIDDEN, NKVHEADS * HEADDIM);
  k_transpose_w<<<dim3((NKVHEADS * HEADDIM) / 32, HIDDEN / 32), 256, 0, stream>>>(Wv, WvT, HIDDEN, NKVHEADS * HEADDIM);
  k_transpose_w<<<dim3(HIDDEN / 32, HIDDEN / 32), 256, 0, stream>>>(Wo, WoT, HIDDEN, HIDDEN);

  // 3) QKV projections (bf16 out)
  k_gemm_bf16<true><<<dim3(HIDDEN / 128, M / 128), 256, 0, stream>>>(Xb, WqT, Qb, M, HIDDEN, HIDDEN);
  k_gemm_bf16<true><<<dim3((NKVHEADS * HEADDIM) / 128, M / 128), 256, 0, stream>>>(Xb, WkT, Kb, M, NKVHEADS * HEADDIM, HIDDEN);
  k_gemm_bf16<true><<<dim3((NKVHEADS * HEADDIM) / 128, M / 128), 256, 0, stream>>>(Xb, WvT, Vb, M, NKVHEADS * HEADDIM, HIDDEN);

  // 4) RoPE on Q and K (in place)
  {
    size_t totq = (size_t)M * NHEADS * 64;
    k_rope<<<dim3(totq / 256), 256, 0, stream>>>(Qb, pos, 5, totq);
    size_t totk = (size_t)M * NKVHEADS * 64;
    k_rope<<<dim3(totk / 256), 256, 0, stream>>>(Kb, pos, 3, totk);
  }

  // 5) flash attention -> Ab (bf16)
  k_attn<<<dim3(SEQ / 64, NHEADS, BATCH), 128, 0, stream>>>(Qb, Kb, Vb, amask, Ab);

  // 6) output projection (fp32 out)
  k_gemm_bf16<false><<<dim3(HIDDEN / 128, M / 128), 256, 0, stream>>>(Ab, WoT, d_out, M, HIDDEN, HIDDEN);
}
